// Cascade_3813930958983
// MI455X (gfx1250) — matrix operand order repaired and checked by reading, not compiled, not run on hardware
//
#include <hip/hip_runtime.h>

typedef __attribute__((ext_vector_type(16))) __bf16 v16bf;
typedef __attribute__((ext_vector_type(8)))  float  v8f;
typedef __attribute__((ext_vector_type(4)))  unsigned int u32x4;
typedef __attribute__((ext_vector_type(4)))  float  f32x4;

constexpr int kB    = 65536;
constexpr int kIn   = 512;
constexpr int kN    = 256;
constexpr int kOut  = 64;
constexpr int kLinW = kN + kOut;      // 320
constexpr int kTileB = 32;
constexpr int kNTLin = kLinW / 16;    // 20 N-tiles for lin GEMM
constexpr int kKTLin = kIn / 32;      // 16 K-tiles for lin GEMM
constexpr int kNTOut = kOut / 16;     // 4
constexpr int kKTOut = kN / 32;       // 8
constexpr int kBlocks = kN / 16;      // 16 cascade blocks

// ---- LDS layout (bytes). x-tile region is reused by tanh/base buffers. ----
constexpr int LP  = 321;              // lin row stride (f32), odd -> bank spread
constexpr int XP  = 520;              // x row stride (halves); 1040B = 65*16 (16B aligned)
constexpr int TBP = 264;              // tanh row stride (halves); 528B = 33*16
constexpr int LIN_OFF  = 0;                          // f32 [32][321]  = 41088 B
constexpr int XB_OFF   = LIN_OFF + kTileB * LP * 4;  // bf16 [32][520] = 33280 B
constexpr int TB_OFF   = XB_OFF;                     // alias (x dead after GEMM1)
constexpr int BASE_OFF = TB_OFF + kTileB * TBP * 2;  // f32 [32][16]   = 2048 B
constexpr int SMEM_END_X = XB_OFF + kTileB * XP * 2;
constexpr int SMEM_END_T = BASE_OFF + kTileB * 16 * 4;
constexpr int SMEM_BYTES = SMEM_END_X > SMEM_END_T ? SMEM_END_X : SMEM_END_T; // 74368

__device__ __forceinline__ unsigned short f2bf(float f) {
  unsigned int u = __float_as_uint(f);
  return (unsigned short)((u + 0x7FFFu + ((u >> 16) & 1u)) >> 16);  // RNE
}
__device__ __forceinline__ float bf2f(unsigned short h) {
  return __uint_as_float((unsigned int)h << 16);
}

// A fragment (16-bit, 16x32 MxK) from LDS row-major halves:
// lanes 0-15: halves[0..15]=K..K+15 (row = lane)
// lanes 16-31: K+16..K+31. Two 16B ds loads per lane.
__device__ __forceinline__ v16bf load_a_lds(const unsigned short* base, int strideH,
                                            int m0, int kbase, int lane) {
  int m  = m0 + (lane & 15);
  int hi = lane >> 4;
  const unsigned short* p = base + m * strideH + kbase + 16 * hi;
  union { u32x4 u[2]; v16bf bf; } f;
  f.u[0] = *(const u32x4*)(p);
  f.u[1] = *(const u32x4*)(p + 8);
  return f.bf;
}

// B fragment pre-swizzled in workspace: [tile][lane][16 halves] contiguous.
// (lane L<16: col=L, K=0..15; L>=16: col=L-16, K=16..31)
__device__ __forceinline__ v16bf load_b_ws(const unsigned short* frag, int tileIdx, int lane) {
  const u32x4* p = (const u32x4*)(frag + (((tileIdx << 5) + lane) << 4));
  union { u32x4 u[2]; v16bf bf; } f;
  f.u[0] = p[0];
  f.u[1] = p[1];
  return f.bf;
}

__device__ __forceinline__ v8f wmma_bf16(v16bf a, v16bf b, v8f c) {
  return __builtin_amdgcn_wmma_f32_16x16x32_bf16(false, a, false, b, (short)0, c, false, false);
}

// ---- prep: swizzle weights into B-fragment layout (bf16) in workspace ----
__global__ void prep_wlin(const float* __restrict__ W, unsigned short* __restrict__ dst) {
  int idx = blockIdx.x * 256 + threadIdx.x;
  if (idx >= kIn * kLinW) return;
  int k = idx / kLinW, n = idx % kLinW;
  int kt = k >> 5, kk = k & 31, nt = n >> 4, nn = n & 15;
  int L = nn + ((kk >> 4) << 4), h = kk & 15;
  dst[(((kt * kNTLin + nt) * 32 + L) << 4) + h] = f2bf(W[idx]);
}

// Masked cascade weights: block p, columns c = 16p+cc, element (k=j, n=cc)
// value = W_nl[c-1][j] if j < c else 0  (exact triangular mask baked in)
__global__ void prep_unl(const float* __restrict__ Wnl, unsigned short* __restrict__ dst) {
  int idx = blockIdx.x * 256 + threadIdx.x;
  if (idx >= kBlocks * kN * 16) return;
  int p  = idx >> 12;
  int k  = (idx >> 4) & 255;
  int cc = idx & 15;
  int c  = p * 16 + cc;
  float v = 0.f;
  if (c >= 1 && k < c) v = Wnl[(c - 1) * kN + k];
  int kt = k >> 5, kk = k & 31;
  int L = cc + ((kk >> 4) << 4), h = kk & 15;
  dst[(((p * 8 + kt) * 32 + L) << 4) + h] = f2bf(v);
}

// W_out is [64][256]; B matrix is W_out^T: element (k, n=o) = W_out[o*256+k]
__global__ void prep_wout(const float* __restrict__ W, unsigned short* __restrict__ dst) {
  int idx = blockIdx.x * 256 + threadIdx.x;
  if (idx >= kOut * kN) return;
  int o = idx / kN, k = idx % kN;
  int kt = k >> 5, kk = k & 31, nt = o >> 4, nn = o & 15;
  int L = nn + ((kk >> 4) << 4), h = kk & 15;
  dst[(((kt * kNTOut + nt) * 32 + L) << 4) + h] = f2bf(W[idx]);
}

// ---- fused main kernel: one workgroup per 32 batch rows ----
__global__ void __launch_bounds__(256)
cascade_main(const float* __restrict__ x,
             const float* __restrict__ Wnl,
             const unsigned short* __restrict__ wlinFrag,
             const unsigned short* __restrict__ ufrag,
             const unsigned short* __restrict__ woutFrag,
             float* __restrict__ out) {
  extern __shared__ char smem[];
  float*          lin  = (float*)(smem + LIN_OFF);
  unsigned short* xb   = (unsigned short*)(smem + XB_OFF);
  unsigned short* tb   = (unsigned short*)(smem + TB_OFF);
  float*          base = (float*)(smem + BASE_OFF);

  const int tid  = threadIdx.x;
  const int lane = tid & 31;
  const int wave = tid >> 5;
  const long tile = blockIdx.x;

  // Phase 0: coalesced load + f32->bf16 convert of x tile into LDS
  {
    const f32x4* xg = (const f32x4*)(x + tile * (long)(kTileB * kIn));
    for (int e = tid; e < kTileB * kIn / 4; e += 256) {
      f32x4 v = xg[e];
      int r = e >> 7, c = (e & 127) << 2;
      unsigned short* d = xb + r * XP + c;
      d[0] = f2bf(v[0]); d[1] = f2bf(v[1]); d[2] = f2bf(v[2]); d[3] = f2bf(v[3]);
    }
  }
  __syncthreads();

  // Phase 1: lin[32][320] = x[32][512] @ W_lin  (40 output tiles over 8 waves)
  for (int t5 = 0; t5 < 5; ++t5) {
    int id = wave * 5 + t5;
    int nt = id >> 1, mt = id & 1;
    v8f acc = {};
    for (int kt = 0; kt < kKTLin; ++kt) {
      v16bf a = load_a_lds(xb, XP, mt * 16, kt * 32, lane);
      v16bf b = load_b_ws(wlinFrag, kt * kNTLin + nt, lane);
      acc = wmma_bf16(a, b, acc);
    }
    int n  = nt * 16 + (lane & 15);
    int rb = mt * 16 + ((lane >> 4) << 3);
#pragma unroll
    for (int r = 0; r < 8; ++r) lin[(rb + r) * LP + n] = acc[r];
  }
  __syncthreads();

  // zero tanh buffer (x region is dead now; zeros make masked WMMA terms exact 0)
  for (int e = tid; e < kTileB * TBP; e += 256) tb[e] = 0;
  __syncthreads();

  // Phase 2: blocked cascade. Per 16-neuron block: WMMA pre-dot over all
  // previous neurons (per-column masked weights), then a short serial tail.
  for (int p = 0; p < kBlocks; ++p) {
    base[tid] = 0.f;
    base[tid + 256] = 0.f;
    __syncthreads();

    const int ktmax = (p >> 1) + 1;  // K-tiles covering j <= 16p+14
    {
      int mt = wave & 1;
      v8f acc = {};
      int used = 0;
      for (int kt = (wave >> 1); kt < ktmax; kt += 4) {
        v16bf a = load_a_lds(tb, TBP, mt * 16, kt * 32, lane);
        v16bf b = load_b_ws(ufrag, p * 8 + kt, lane);
        acc = wmma_bf16(a, b, acc);
        used = 1;
      }
      if (used) {
        int n  = lane & 15;
        int rb = mt * 16 + ((lane >> 4) << 3);
#pragma unroll
        for (int r = 0; r < 8; ++r)
          atomicAdd(&base[(rb + r) * 16 + n], acc[r]);  // ds_add_f32 combine
      }
    }
    __syncthreads();

    // serial intra-block recurrence: one lane per batch row (wave 0)
    if (tid < kTileB) {
      const int row = tid;
      unsigned short* trow = tb + row * TBP;
#pragma unroll 1
      for (int cc = 0; cc < 16; ++cc) {
        int c = p * 16 + cc;
        float v;
        if (c == 0) {
          v = lin[row * LP] + 1.f;                       // acc0[:,0] = lin[:,0]+1
        } else {
          v = base[row * 16 + cc] + lin[row * LP + (c - 1)] + 1.f;
          for (int j = p * 16; j < c; ++j)               // intra-block terms
            v += bf2f(trow[j]) * Wnl[(c - 1) * kN + j];  // broadcast load
        }
        trow[c] = f2bf(tanhf(v));
      }
    }
    __syncthreads();
  }

  // Phase 3: out[32][64] = tanh(acc) @ W_out^T + lin[:,256:320] + 1
  {
    int mt = wave & 1, nt = wave >> 1;
    v8f acc = {};
    for (int kt = 0; kt < kKTOut; ++kt) {
      v16bf a = load_a_lds(tb, TBP, mt * 16, kt * 32, lane);
      v16bf b = load_b_ws(woutFrag, kt * kNTOut + nt, lane);
      acc = wmma_bf16(a, b, acc);
    }
    int n  = nt * 16 + (lane & 15);
    int rb = mt * 16 + ((lane >> 4) << 3);
#pragma unroll
    for (int r = 0; r < 8; ++r) {
      int row = rb + r;
      out[(tile * kTileB + row) * (long)kOut + n] =
          acc[r] + lin[row * LP + kN + n] + 1.f;
    }
  }
}

extern "C" void kernel_launch(void* const* d_in, const int* in_sizes, int n_in,
                              void* d_out, int out_size, void* d_ws, size_t ws_size,
                              hipStream_t stream) {
  (void)in_sizes; (void)n_in; (void)out_size; (void)ws_size;
  const float* x    = (const float*)d_in[0];
  const float* Wlin = (const float*)d_in[1];
  const float* Wnl  = (const float*)d_in[2];
  const float* Wout = (const float*)d_in[3];
  float* out = (float*)d_out;

  // workspace: bf16 fragment-swizzled weights (total ~480 KB, L2-resident)
  unsigned short* wlinFrag = (unsigned short*)d_ws;               // 512*320 halves
  unsigned short* ufrag    = wlinFrag + kIn * kLinW;              // 16*8*32*16 halves
  unsigned short* woutFrag = ufrag + kBlocks * 8 * 32 * 16;       // 256*64 halves

  prep_wlin<<<(kIn * kLinW + 255) / 256, 256, 0, stream>>>(Wlin, wlinFrag);
  prep_unl<<<(kBlocks * kN * 16 + 255) / 256, 256, 0, stream>>>(Wnl, ufrag);
  prep_wout<<<(kOut * kN + 255) / 256, 256, 0, stream>>>(Wout, woutFrag);

  (void)hipFuncSetAttribute(reinterpret_cast<const void*>(cascade_main),
                            hipFuncAttributeMaxDynamicSharedMemorySize, SMEM_BYTES);
  cascade_main<<<kB / kTileB, 256, SMEM_BYTES, stream>>>(
      x, Wnl, wlinFrag, ufrag, woutFrag, out);
}
